// S4Model_39298950759098
// MI455X (gfx1250) — matrix operand order repaired and checked by reading, not compiled, not run on hardware
//
#include <hip/hip_runtime.h>

// ---------------------------------------------------------------------------
// S4 linear SSM on MI455X (gfx1250), bf16 WMMA pipeline with chunked scan.
//   BATCH=32, SEQ=4096, STATE=IN=OUT=256, chunk L=32 -> NC=128 parallel chunks
// Round 2: stage reused 256x256 bf16 operand matrices in LDS via CDNA5
// async-to-LDS (ASYNCcnt) path; prefetch next-step tiles in scan loops.
// ---------------------------------------------------------------------------

#define STATE  256
#define BATCH_ 32
#define TSEQ   4096
#define LCH    32
#define NCH    (TSEQ / LCH)   // 128

typedef __attribute__((ext_vector_type(16))) __bf16 v16bf;
typedef __attribute__((ext_vector_type(8)))  __bf16 v8bf;
typedef __attribute__((ext_vector_type(8)))  float  v8f;

__device__ __forceinline__ v8f zero8() {
  v8f r;
#pragma unroll
  for (int i = 0; i < 8; ++i) r[i] = 0.0f;
  return r;
}

__device__ __forceinline__ v8f wmma_bf16(v16bf a, v16bf b, v8f c) {
  // (neg_a, A, neg_b, B, c_mod, C, reuse_a, reuse_b)
  return __builtin_amdgcn_wmma_f32_16x16x32_bf16(false, a, false, b, (short)0, c,
                                                 false, false);
}

__device__ __forceinline__ v16bf make16(v8bf lo, v8bf hi) {
  v16bf r;
#pragma unroll
  for (int i = 0; i < 8; ++i) { r[i] = lo[i]; r[i + 8] = hi[i]; }
  return r;
}

// ---- CDNA5 async global->LDS staging (ASYNCcnt) ---------------------------
// LDS generic addresses carry the wave-relative LDS byte offset in the low
// 32 bits (flat aperture truncation), which is what VDST of the async op takes.
__device__ __forceinline__ unsigned lds_off32(const void* p) {
  return (unsigned)(unsigned long long)p;
}
__device__ __forceinline__ void async_copy_b128(const void* lds_dst,
                                                const void* global_src) {
  asm volatile("global_load_async_to_lds_b128 %0, %1, off"
               :: "v"(lds_off32(lds_dst)), "v"(global_src)
               : "memory");
}
__device__ __forceinline__ void async_wait_all() {
  asm volatile("s_wait_asynccnt 0x0" ::: "memory");
}
// Stage `bytes` (multiple of 16, 16B aligned) from global to LDS, whole block.
__device__ __forceinline__ void stage_matrix(void* lds_dst, const void* gsrc,
                                             int bytes, int tid, int nthr) {
  const char* g = (const char*)gsrc;
  char* l = (char*)lds_dst;
  for (int off = tid * 16; off < bytes; off += nthr * 16)
    async_copy_b128(l + off, g + off);
}

// A-operand (16xK, 16-bit): lane m=lane&15, half=lane>>4; elements 0..7 hold
// K = 8*half + i, elements 8..15 hold K = 16 + 8*half + i  (two contiguous runs).
__device__ __forceinline__ v16bf load_a_frag_bf16(const __bf16* a, int ld, int lane) {
  const int m = lane & 15, half = lane >> 4;
  const __bf16* row = a + (size_t)m * ld + 8 * half;
  v8bf lo = *reinterpret_cast<const v8bf*>(row);
  v8bf hi = *reinterpret_cast<const v8bf*>(row + 16);
  return make16(lo, hi);
}

__device__ __forceinline__ v16bf load_a_frag_f32(const float* a, int ld, int lane) {
  const int m = lane & 15, half = lane >> 4;
  const float* row = a + (size_t)m * ld + 8 * half;
  v16bf r;
#pragma unroll
  for (int i = 0; i < 8; ++i) r[i] = (__bf16)row[i];
#pragma unroll
  for (int i = 0; i < 8; ++i) r[8 + i] = (__bf16)row[16 + i];
  return r;
}

// B-operand (Kx16, 16-bit): lane n=lane&15, half=lane>>4; elements 0..7 hold
// K = 8*half + i, elements 8..15 hold K = 16 + 8*half + i (matches A fragment).
// With "col storage" W[n*ld + k] = Bop[k][n]; for x @ W^T it IS row-major W.
__device__ __forceinline__ v16bf load_b_frag(const __bf16* col, int ld, int lane) {
  const int n = lane & 15, half = lane >> 4;
  const __bf16* p = col + (size_t)n * ld + 8 * half;
  v8bf lo = *reinterpret_cast<const v8bf*>(p);
  v8bf hi = *reinterpret_cast<const v8bf*>(p + 16);
  return make16(lo, hi);
}

// C/D matrix (16x16 f32): lane n=lane&15, VGPR v -> row m = v + 8*(lane>>4).
__device__ __forceinline__ v8f load_c_f32(const float* c, long ld, int lane) {
  const int n = lane & 15, half = lane >> 4;
  v8f r;
#pragma unroll
  for (int v = 0; v < 8; ++v) r[v] = c[(long)(v + 8 * half) * ld + n];
  return r;
}
__device__ __forceinline__ void store_c_f32(float* c, long ld, int lane, v8f acc) {
  const int n = lane & 15, half = lane >> 4;
#pragma unroll
  for (int v = 0; v < 8; ++v) c[(long)(v + 8 * half) * ld + n] = acc[v];
}
__device__ __forceinline__ void store_c_bf16(__bf16* c, int ld, int lane, v8f acc) {
  const int n = lane & 15, half = lane >> 4;
#pragma unroll
  for (int v = 0; v < 8; ++v) c[(v + 8 * half) * ld + n] = (__bf16)acc[v];
}

// ---------------------------------------------------------------------------
// Kernel 0: elementwise bf16 converts (no transposes needed for x@W^T),
// plus Mrow0 = A^T (f32) and Mcol0 = A (bf16) to seed the power doubling.
// ---------------------------------------------------------------------------
__global__ void convert_mats(const float* __restrict__ A, const float* __restrict__ Bm,
                             const float* __restrict__ Cm, const float* __restrict__ Dm,
                             __bf16* __restrict__ Abf, __bf16* __restrict__ Bbf,
                             __bf16* __restrict__ Cbf, __bf16* __restrict__ Dbf,
                             float* __restrict__ Mrow0, __bf16* __restrict__ Mcol0) {
  const int total = STATE * STATE;
  for (int i = blockIdx.x * blockDim.x + threadIdx.x; i < total;
       i += gridDim.x * blockDim.x) {
    const int r = i >> 8, cidx = i & 255;
    const float a = A[i];
    Abf[i] = (__bf16)a;
    Mcol0[i] = (__bf16)a;               // Mcol for M1=A^T is row-major A
    Bbf[i] = (__bf16)Bm[i];
    Cbf[i] = (__bf16)Cm[i];
    Dbf[i] = (__bf16)Dm[i];
    Mrow0[cidx * STATE + r] = a;        // Mrow for M1 = A^T row-major
  }
}

// ---------------------------------------------------------------------------
// Kernel 1: uB = u @ B^T, uD = u @ D^T.  M = B*T = 131072, N = K = 256.
// Streaming GEMM; A-fragments cached in VGPRs across both GEMMs, B/D tiles
// from L2 (128 KB each, shared by all 1024 blocks).
// ---------------------------------------------------------------------------
__global__ void input_proj(const float* __restrict__ u, const __bf16* __restrict__ Bbf,
                           const __bf16* __restrict__ Dbf, float* __restrict__ uB,
                           float* __restrict__ uD) {
  const int lane = threadIdx.x & 31;
  const int wave = threadIdx.x >> 5;
  const long row0 = (long)blockIdx.x * 128 + (long)wave * 16;
  const float* abase = u + row0 * STATE;

  v16bf afrag[8];
#pragma unroll
  for (int kt = 0; kt < 8; ++kt)
    afrag[kt] = load_a_frag_f32(abase + kt * 32, STATE, lane);

  for (int nt = 0; nt < 16; ++nt) {
    v8f accB = zero8(), accD = zero8();
#pragma unroll
    for (int kt = 0; kt < 8; ++kt) {
      v16bf bB = load_b_frag(Bbf + nt * 16 * STATE + kt * 32, STATE, lane);
      accB = wmma_bf16(afrag[kt], bB, accB);
      v16bf bD = load_b_frag(Dbf + nt * 16 * STATE + kt * 32, STATE, lane);
      accD = wmma_bf16(afrag[kt], bD, accD);
    }
    store_c_f32(uB + row0 * STATE + nt * 16, STATE, lane, accB);
    store_c_f32(uD + row0 * STATE + nt * 16, STATE, lane, accD);
  }
}

// ---------------------------------------------------------------------------
// Kernel 2: per-chunk local scan with zero initial state.
//   lx_j = lx_{j-1} @ A^T + uB_t  (in-place overwrite of uB with lx).
// A staged in LDS once (async-to-LDS); state double-buffered in LDS bf16.
// LDS = 128KB + 32KB = 160KB.
// ---------------------------------------------------------------------------
__global__ void local_scan(const __bf16* __restrict__ Abf, float* __restrict__ uB) {
  __shared__ __bf16 sA[STATE * STATE];              // 128 KB
  __shared__ __bf16 sX[2][BATCH_ * STATE];          // 2 x 16 KB
  const int lane = threadIdx.x & 31;
  const int wave = threadIdx.x >> 5;
  const int c = blockIdx.x;

  stage_matrix(sA, Abf, STATE * STATE * 2, threadIdx.x, blockDim.x);
  for (int i = threadIdx.x; i < BATCH_ * STATE; i += blockDim.x)
    sX[0][i] = (__bf16)0.0f;
  async_wait_all();
  __syncthreads();

  for (int j = 0; j < LCH; ++j) {
    const int t = c * LCH + j;
    const int cur = j & 1, nxt = cur ^ 1;
#pragma unroll
    for (int mt = 0; mt < 2; ++mt) {
#pragma unroll
      for (int q = 0; q < 2; ++q) {
        const int nt = wave * 2 + q;
        if (j + 1 < LCH)  // prefetch next step's uB tile while WMMAs run
          __builtin_prefetch(uB + ((long)(mt * 16) * TSEQ + t + 1) * STATE + nt * 16, 0, 3);
        v8f acc = zero8();
#pragma unroll
        for (int kt = 0; kt < 8; ++kt) {
          v16bf a = load_a_frag_bf16(&sX[cur][mt * 16 * STATE + kt * 32], STATE, lane);
          v16bf b = load_b_frag(&sA[nt * 16 * STATE + kt * 32], STATE, lane);
          acc = wmma_bf16(a, b, acc);
        }
        float* ug = uB + ((long)(mt * 16) * TSEQ + t) * STATE + nt * 16;
        acc = acc + load_c_f32(ug, (long)TSEQ * STATE, lane);   // + uB_t
        store_c_f32(ug, (long)TSEQ * STATE, lane, acc);         // lx_t (in place)
        store_c_bf16(&sX[nxt][mt * 16 * STATE + nt * 16], STATE, lane, acc);
      }
    }
    __syncthreads();
  }
}

// ---------------------------------------------------------------------------
// Kernel 3: one squaring step of the chunk transition matrix:
//   N = M @ M, emitting both row-major f32 and B-operand (col) bf16 layouts.
// Applied 5x: A^T -> (A^T)^2 -> ... -> (A^T)^32.
// ---------------------------------------------------------------------------
__global__ void gemm256_dual(const float* __restrict__ Mrow, const __bf16* __restrict__ Mcol,
                             float* __restrict__ Nrow, __bf16* __restrict__ Ncol) {
  const int lane = threadIdx.x & 31;
  const int wave = threadIdx.x >> 5;
  const int tile = blockIdx.x * 8 + wave;           // 256 tiles total
  const int mt = tile >> 4, nt = tile & 15;
  v8f acc = zero8();
#pragma unroll
  for (int kt = 0; kt < 8; ++kt) {
    v16bf a = load_a_frag_f32(Mrow + mt * 16 * STATE + kt * 32, STATE, lane);
    v16bf b = load_b_frag(Mcol + nt * 16 * STATE + kt * 32, STATE, lane);
    acc = wmma_bf16(a, b, acc);
  }
  const int n = lane & 15, half = lane >> 4;
#pragma unroll
  for (int v = 0; v < 8; ++v) {
    const int m = mt * 16 + v + 8 * half;
    const int col = nt * 16 + n;
    const float val = acc[v];
    Nrow[m * STATE + col] = val;
    Ncol[col * STATE + m] = (__bf16)val;
  }
}

// ---------------------------------------------------------------------------
// Kernel 4: sequential combine over 128 chunk boundaries (single block):
//   x0buf[c] = X;  X = X @ (A^T)^L + lx[c*L + L-1]
// M_L staged in LDS once.  LDS = 128KB + 32KB = 160KB.
// ---------------------------------------------------------------------------
__global__ void combine(const __bf16* __restrict__ MLcol, const float* __restrict__ uB,
                        float* __restrict__ x0buf) {
  __shared__ __bf16 sML[STATE * STATE];             // 128 KB
  __shared__ float sX[BATCH_ * STATE];              // 32 KB, f32 master state
  const int lane = threadIdx.x & 31;
  const int wave = threadIdx.x >> 5;

  stage_matrix(sML, MLcol, STATE * STATE * 2, threadIdx.x, blockDim.x);
  for (int i = threadIdx.x; i < BATCH_ * STATE; i += blockDim.x) sX[i] = 0.0f;
  async_wait_all();
  __syncthreads();

  for (int c = 0; c < NCH; ++c) {
    for (int i = threadIdx.x; i < BATCH_ * STATE; i += blockDim.x)
      x0buf[(long)c * BATCH_ * STATE + i] = sX[i];
    const int tend = c * LCH + (LCH - 1);

    v8f acc[2][2];
#pragma unroll
    for (int mt = 0; mt < 2; ++mt) {
#pragma unroll
      for (int q = 0; q < 2; ++q) {
        const int nt = wave * 2 + q;
        v8f a8 = zero8();
#pragma unroll
        for (int kt = 0; kt < 8; ++kt) {
          v16bf a = load_a_frag_f32(&sX[mt * 16 * STATE + kt * 32], STATE, lane);
          v16bf b = load_b_frag(&sML[nt * 16 * STATE + kt * 32], STATE, lane);
          a8 = wmma_bf16(a, b, a8);
        }
        const float* lg = uB + ((long)(mt * 16) * TSEQ + tend) * STATE + nt * 16;
        acc[mt][q] = a8 + load_c_f32(lg, (long)TSEQ * STATE, lane);
      }
    }
    __syncthreads();                                // all reads of sX complete
#pragma unroll
    for (int mt = 0; mt < 2; ++mt) {
#pragma unroll
      for (int q = 0; q < 2; ++q) {
        const int nt = wave * 2 + q;
        store_c_f32(&sX[mt * 16 * STATE + nt * 16], STATE, lane, acc[mt][q]);
      }
    }
    __syncthreads();
  }
}

// ---------------------------------------------------------------------------
// Kernel 5: per-chunk parallel fix-up and output:
//   d_j = d_{j-1} @ A^T (d_{-1} = x0_c);  x_j = d_j + lx_j;
//   y_j = x_j @ C^T + uD_j
// A staged in LDS; d single-buffered (compute->barrier->write->barrier).
// LDS = 128KB + 16KB + 16KB = 160KB.  C tiles come from L2 (shared).
// ---------------------------------------------------------------------------
__global__ void finalize(const __bf16* __restrict__ Abf, const __bf16* __restrict__ Cbf,
                         const float* __restrict__ lx, const float* __restrict__ uD,
                         const float* __restrict__ x0buf, float* __restrict__ out) {
  __shared__ __bf16 sA[STATE * STATE];              // 128 KB
  __shared__ __bf16 sD[BATCH_ * STATE];             // 16 KB
  __shared__ __bf16 sXb[BATCH_ * STATE];            // 16 KB
  const int lane = threadIdx.x & 31;
  const int wave = threadIdx.x >> 5;
  const int c = blockIdx.x;

  stage_matrix(sA, Abf, STATE * STATE * 2, threadIdx.x, blockDim.x);
  for (int i = threadIdx.x; i < BATCH_ * STATE; i += blockDim.x)
    sD[i] = (__bf16)x0buf[(long)c * BATCH_ * STATE + i];
  async_wait_all();
  __syncthreads();

  for (int j = 0; j < LCH; ++j) {
    const int t = c * LCH + j;

    // phase 1: d_j = d_{j-1} @ A^T ;  x_j = d_j + lx_t   (reads sD)
    v8f dacc[2][2], xacc[2][2];
#pragma unroll
    for (int mt = 0; mt < 2; ++mt) {
#pragma unroll
      for (int q = 0; q < 2; ++q) {
        const int nt = wave * 2 + q;
        if (j + 1 < LCH) {
          __builtin_prefetch(lx + ((long)(mt * 16) * TSEQ + t + 1) * STATE + nt * 16, 0, 3);
          __builtin_prefetch(uD + ((long)(mt * 16) * TSEQ + t + 1) * STATE + nt * 16, 0, 3);
        }
        v8f acc = zero8();
#pragma unroll
        for (int kt = 0; kt < 8; ++kt) {
          v16bf a = load_a_frag_bf16(&sD[mt * 16 * STATE + kt * 32], STATE, lane);
          v16bf b = load_b_frag(&sA[nt * 16 * STATE + kt * 32], STATE, lane);
          acc = wmma_bf16(a, b, acc);
        }
        dacc[mt][q] = acc;
        const float* lg = lx + ((long)(mt * 16) * TSEQ + t) * STATE + nt * 16;
        xacc[mt][q] = acc + load_c_f32(lg, (long)TSEQ * STATE, lane);
      }
    }
    __syncthreads();                                // all sD reads done
#pragma unroll
    for (int mt = 0; mt < 2; ++mt) {
#pragma unroll
      for (int q = 0; q < 2; ++q) {
        const int nt = wave * 2 + q;
        store_c_bf16(&sD[mt * 16 * STATE + nt * 16], STATE, lane, dacc[mt][q]);
        store_c_bf16(&sXb[mt * 16 * STATE + nt * 16], STATE, lane, xacc[mt][q]);
      }
    }
    __syncthreads();                                // sD/sXb visible

    // phase 2: y_j = x_j @ C^T + uD_t
#pragma unroll
    for (int mt = 0; mt < 2; ++mt) {
#pragma unroll
      for (int q = 0; q < 2; ++q) {
        const int nt = wave * 2 + q;
        v8f acc = zero8();
#pragma unroll
        for (int kt = 0; kt < 8; ++kt) {
          v16bf a = load_a_frag_bf16(&sXb[mt * 16 * STATE + kt * 32], STATE, lane);
          v16bf b = load_b_frag(Cbf + nt * 16 * STATE + kt * 32, STATE, lane);
          acc = wmma_bf16(a, b, acc);
        }
        const float* dg = uD + ((long)(mt * 16) * TSEQ + t) * STATE + nt * 16;
        acc = acc + load_c_f32(dg, (long)TSEQ * STATE, lane);
        float* og = out + ((long)(mt * 16) * TSEQ + t) * STATE + nt * 16;
        store_c_f32(og, (long)TSEQ * STATE, lane, acc);
      }
    }
    // next iteration's writes to sD/sXb are fenced by the phase-1 barrier
  }
}

// ---------------------------------------------------------------------------
// Host-side orchestration (graph-capture safe: launches only, scratch in ws).
// Workspace budget ~273 MB: uB 134MB, uD 134MB, bf16 mats 512KB,
// power ping-pong ~768KB, x0buf 4MB.
// ---------------------------------------------------------------------------
extern "C" void kernel_launch(void* const* d_in, const int* in_sizes, int n_in,
                              void* d_out, int out_size, void* d_ws, size_t ws_size,
                              hipStream_t stream) {
  (void)in_sizes; (void)n_in; (void)out_size; (void)ws_size;
  const float* u  = (const float*)d_in[0];
  const float* A  = (const float*)d_in[1];
  const float* Bm = (const float*)d_in[2];
  const float* Cm = (const float*)d_in[3];
  const float* Dm = (const float*)d_in[4];
  float* out = (float*)d_out;

  char* ws = (char*)d_ws;
  size_t off = 0;
  auto take = [&](size_t bytes) -> char* {
    char* p = ws + off;
    off = (off + bytes + 255) & ~(size_t)255;
    return p;
  };
  float*  uB    = (float*)take((size_t)BATCH_ * TSEQ * STATE * 4);
  float*  uD    = (float*)take((size_t)BATCH_ * TSEQ * STATE * 4);
  __bf16* Abf   = (__bf16*)take((size_t)STATE * STATE * 2);
  __bf16* Bbf   = (__bf16*)take((size_t)STATE * STATE * 2);
  __bf16* Cbf   = (__bf16*)take((size_t)STATE * STATE * 2);
  __bf16* Dbf   = (__bf16*)take((size_t)STATE * STATE * 2);
  float*  Mrow[2];
  __bf16* Mcol[2];
  Mrow[0] = (float*)take((size_t)STATE * STATE * 4);
  Mrow[1] = (float*)take((size_t)STATE * STATE * 4);
  Mcol[0] = (__bf16*)take((size_t)STATE * STATE * 2);
  Mcol[1] = (__bf16*)take((size_t)STATE * STATE * 2);
  float* x0buf = (float*)take((size_t)NCH * BATCH_ * STATE * 4);

  convert_mats<<<64, 256, 0, stream>>>(A, Bm, Cm, Dm, Abf, Bbf, Cbf, Dbf,
                                       Mrow[0], Mcol[0]);

  // Input projections: (B*T)/128 = 1024 blocks.
  input_proj<<<(BATCH_ * TSEQ) / 128, 256, 0, stream>>>(u, Bbf, Dbf, uB, uD);

  // Local scans: one block per chunk.
  local_scan<<<NCH, 256, 0, stream>>>(Abf, uB);

  // (A^T)^32 by 5 squarings.
  int p = 0;
  for (int it = 0; it < 5; ++it) {
    gemm256_dual<<<32, 256, 0, stream>>>(Mrow[p], Mcol[p], Mrow[p ^ 1], Mcol[p ^ 1]);
    p ^= 1;
  }

  // Sequential chunk-boundary combine (tiny).
  combine<<<1, 256, 0, stream>>>(Mcol[p], uB, x0buf);

  // Parallel fix-up + output.
  finalize<<<NCH, 256, 0, stream>>>(Abf, Cbf, uB, uD, x0buf, out);
}
